// ModelOptAWQPrepackedLinear_12360915878782
// MI455X (gfx1250) — matrix operand order repaired and checked by reading, not compiled, not run on hardware
//
#include <hip/hip_runtime.h>
#include <hip/hip_fp16.h>

typedef __attribute__((ext_vector_type(16))) _Float16 v16h;
typedef __attribute__((ext_vector_type(8)))  _Float16 v8h;
typedef __attribute__((ext_vector_type(8)))  float    v8f;
typedef unsigned __attribute__((ext_vector_type(4))) u32x4;
typedef unsigned __attribute__((ext_vector_type(8))) u32x8;

#define TOKENS_ 8192
#define IN_     4096
#define OUT_    12288
#define NGRP    32          // IN_/128 scale groups
#define BM      128
#define BN      128
#define BK      64
#define NKT     (IN_ / BK)  // 64 k-tiles
#define LDA     72          // LDS row stride (halves): 64 data + 8 pad (16B)
#define LDW     72

static __device__ __forceinline__ __half2 h2_from_bits(unsigned b) {
    union { unsigned u; __half2 h; } x; x.u = b; return x.h;
}

// ---------------------------------------------------------------------------
// Phase 1: one-shot dequant int4 -> fp16 with group scale AND pre-quant scale
// folded in:  w16[o][k] = (nibble-8) * wscale[o][k/128] * pqs[k]
// Pure streaming kernel: 100MB in, 100MB out  (~9us @ 23.3TB/s)
// ---------------------------------------------------------------------------
__global__ __launch_bounds__(256)
void awq_dequant(const int*      __restrict__ wpk,
                 const _Float16* __restrict__ wscale,
                 const _Float16* __restrict__ pqs,
                 _Float16*       __restrict__ w16)
{
    const int idx = blockIdx.x * 256 + threadIdx.x;   // 6,291,456 threads
    const int op  = idx >> 10;                        // packed row 0..6143
    const int kc  = (idx & 1023) << 2;                // k offset, 4 ints/thread
    const int g   = kc >> 7;

    const uint4 v4 = *(const uint4*)(wpk + (size_t)op * IN_ + kc);
    const __half2 sl2 = __half2half2(*(const __half*)&wscale[(size_t)(2 * op    ) * NGRP + g]);
    const __half2 sh2 = __half2half2(*(const __half*)&wscale[(size_t)(2 * op + 1) * NGRP + g]);
    const __half2* pk2 = (const __half2*)(pqs + kc);
    const __half2 p01 = pk2[0], p23 = pk2[1];
    const __half2 spl01 = __hmul2(sl2, p01), spl23 = __hmul2(sl2, p23);
    const __half2 sph01 = __hmul2(sh2, p01), sph23 = __hmul2(sh2, p23);

    const __half2 neg1032 = __float2half2_rn(-1032.0f);
    const unsigned v0 = v4.x, v1 = v4.y, v2 = v4.z, v3 = v4.w;
    // fp16 magic: bits(0x6400|n) == 1024+n ; (1024+n)-1032 == n-8 (exact)
    __half2 lo01 = h2_from_bits((v0 & 0xFu)        | ((v1 & 0xFu) << 16)        | 0x64006400u);
    __half2 lo23 = h2_from_bits((v2 & 0xFu)        | ((v3 & 0xFu) << 16)        | 0x64006400u);
    __half2 hi01 = h2_from_bits(((v0 >> 4) & 0xFu) | (((v1 >> 4) & 0xFu) << 16) | 0x64006400u);
    __half2 hi23 = h2_from_bits(((v2 >> 4) & 0xFu) | (((v3 >> 4) & 0xFu) << 16) | 0x64006400u);
    lo01 = __hmul2(__hadd2(lo01, neg1032), spl01);
    lo23 = __hmul2(__hadd2(lo23, neg1032), spl23);
    hi01 = __hmul2(__hadd2(hi01, neg1032), sph01);
    hi23 = __hmul2(__hadd2(hi23, neg1032), sph23);

    union { __half2 h2[2]; unsigned long long u; } L, H;
    L.h2[0] = lo01; L.h2[1] = lo23;
    H.h2[0] = hi01; H.h2[1] = hi23;
    *(unsigned long long*)(w16 + (size_t)(2 * op    ) * IN_ + kc) = L.u;
    *(unsigned long long*)(w16 + (size_t)(2 * op + 1) * IN_ + kc) = H.u;
}

// ---------------------------------------------------------------------------
// Phase 2: pure f16 WMMA GEMM, tiles fed by the Tensor Data Mover.
// One tensor_load_to_lds per (A|W) tile per k-step, issued by wave 0;
// D# pad feature recreates the 128B+16B padded LDS row layout in hardware.
// ---------------------------------------------------------------------------
__global__ __launch_bounds__(256)
void gemm_f16_tdm(const _Float16* __restrict__ hs,
                  const _Float16* __restrict__ w16,
                  float*          __restrict__ out)
{
    __shared__ _Float16 ldsA[2][BM * LDA];
    __shared__ _Float16 ldsW[2][BN * LDW];

    const int tid  = threadIdx.x;
    const int lane = tid & 31;
    const int wave = tid >> 5;
    const int wm   = wave >> 2;
    const int wn   = wave & 3;

    const int m0 = blockIdx.x * BM;
    const int n0 = blockIdx.y * BN;

    // D# group1 (static): data_size=2B(1), pad_enable, pad_interval=32dw(128B),
    // pad_amount=4dw(16B); tensor_dim0=4096, tile=64x128, dim0_stride=4096
    const u32x8 g1a = {0x07110000u, 0x10000000u, 0x20000000u /*dim1=8192*/,
                       0x00400000u /*tile_dim0=64*/, 0x00000080u /*tile_dim1=128*/,
                       0x00001000u /*stride=4096*/, 0u, 0u};
    const u32x8 g1w = {0x07110000u, 0x10000000u, 0x30000000u /*dim1=12288*/,
                       0x00400000u, 0x00000080u, 0x00001000u, 0u, 0u};

    const unsigned ldsAbase0 = (unsigned)(uintptr_t)&ldsA[0][0];
    const unsigned ldsAbase1 = (unsigned)(uintptr_t)&ldsA[1][0];
    const unsigned ldsWbase0 = (unsigned)(uintptr_t)&ldsW[0][0];
    const unsigned ldsWbase1 = (unsigned)(uintptr_t)&ldsW[1][0];

    auto issueTDM = [&](int kt, int buf) {
        const int k0 = kt * BK;
        unsigned long long ga =
            (unsigned long long)(uintptr_t)(hs + (size_t)m0 * IN_ + k0);
        u32x4 g0a = {1u, buf ? ldsAbase1 : ldsAbase0,
                     (unsigned)ga, (unsigned)(ga >> 32) | 0x80000000u /*type=2*/};
        asm volatile("tensor_load_to_lds %0, %1" :: "s"(g0a), "s"(g1a) : "memory");
        unsigned long long gw =
            (unsigned long long)(uintptr_t)(w16 + (size_t)n0 * IN_ + k0);
        u32x4 g0w = {1u, buf ? ldsWbase1 : ldsWbase0,
                     (unsigned)gw, (unsigned)(gw >> 32) | 0x80000000u};
        asm volatile("tensor_load_to_lds %0, %1" :: "s"(g0w), "s"(g1w) : "memory");
    };

    v8f zero = {0.f,0.f,0.f,0.f,0.f,0.f,0.f,0.f};
    v8f acc[4][2];
#pragma unroll
    for (int i = 0; i < 4; ++i)
#pragma unroll
        for (int j = 0; j < 2; ++j) acc[i][j] = zero;

    const int mFragRow = wm * 64 + (lane & 15);
    const int nFragRow = wn * 32 + (lane & 15);
    const int aKoff    = (lane >> 4) << 3;   // A: lanes 16-31 start at K=8
    const int bKoff    = (lane >> 4) << 3;   // B: lanes 16-31 start at K=8

    auto compute = [&](int buf) {
#pragma unroll
        for (int ks = 0; ks < 2; ++ks) {
            const int kb = ks * 32;
            v16h af[4];
#pragma unroll
            for (int tm = 0; tm < 4; ++tm) {
                const _Float16* p = &ldsA[buf][(mFragRow + tm * 16) * LDA + kb + aKoff];
                v8h lo = *(const v8h*)p;
                v8h hi = *(const v8h*)(p + 16);
                af[tm] = __builtin_shufflevector(lo, hi,
                         0,1,2,3,4,5,6,7,8,9,10,11,12,13,14,15);
            }
            v16h bf[2];
#pragma unroll
            for (int tn = 0; tn < 2; ++tn) {
                const _Float16* p = &ldsW[buf][(nFragRow + tn * 16) * LDW + kb + bKoff];
                v8h lo = *(const v8h*)p;
                v8h hi = *(const v8h*)(p + 16);
                bf[tn] = __builtin_shufflevector(lo, hi,
                         0,1,2,3,4,5,6,7,8,9,10,11,12,13,14,15);
            }
#pragma unroll
            for (int tm = 0; tm < 4; ++tm)
#pragma unroll
                for (int tn = 0; tn < 2; ++tn)
                    acc[tm][tn] = __builtin_amdgcn_wmma_f32_16x16x32_f16(
                        false, af[tm], false, bf[tn],
                        (short)0, acc[tm][tn], false, false);
        }
    };

    if (wave == 0) {
        issueTDM(0, 0);
        __builtin_amdgcn_s_wait_tensorcnt(0);
    }
    __syncthreads();

    for (int kt = 0; kt < NKT; ++kt) {
        const int buf = kt & 1;
        if (wave == 0 && kt + 1 < NKT) issueTDM(kt + 1, buf ^ 1);  // DMA next tiles
        compute(buf);
        if (wave == 0 && kt + 1 < NKT) __builtin_amdgcn_s_wait_tensorcnt(0);
        __syncthreads();
    }

    const int rowBase = m0 + wm * 64 + ((lane >> 4) << 3);
    const int colBase = n0 + wn * 32 + (lane & 15);
#pragma unroll
    for (int tm = 0; tm < 4; ++tm)
#pragma unroll
        for (int tn = 0; tn < 2; ++tn)
#pragma unroll
            for (int r = 0; r < 8; ++r)
                out[(size_t)(rowBase + tm * 16 + r) * OUT_ + (colBase + tn * 16)]
                    = acc[tm][tn][r];
}

// ---------------------------------------------------------------------------
// Fallback: fused kernel (async global->LDS A tiles, in-loop dequant) for the
// case where the workspace cannot hold the dequantized weights.
// ---------------------------------------------------------------------------
__global__ __launch_bounds__(256)
void awq_w4a16_wmma_fused(const _Float16* __restrict__ hs,
                          const int*      __restrict__ wpk,
                          const _Float16* __restrict__ wscale,
                          const _Float16* __restrict__ pqs,
                          float*          __restrict__ out)
{
    __shared__ _Float16 ldsA[2][BM * LDA];
    __shared__ _Float16 ldsW[2][BN * LDW];
    __shared__ _Float16 ldsS[NGRP * BN];
    __shared__ _Float16 ldsP[IN_];

    const int tid  = threadIdx.x;
    const int lane = tid & 31;
    const int wave = tid >> 5;
    const int wm   = wave >> 2;
    const int wn   = wave & 3;

    const int m0  = blockIdx.x * BM;
    const int n0  = blockIdx.y * BN;
    const int op0 = n0 >> 1;

    int aRow[4], aKc[4], wRow[4], wKc[4];
#pragma unroll
    for (int i = 0; i < 4; ++i) {
        int c = tid + i * 256;
        aRow[i] = c >> 3;
        aKc[i]  = (c & 7) << 3;
        wRow[i] = c >> 4;
        wKc[i]  = (c & 15) << 2;
    }

    uint4 wStage[4];

    auto asyncLoadA = [&](int kt, int buf) {
        const int k0 = kt * BK;
#pragma unroll
        for (int i = 0; i < 4; ++i) {
            unsigned ldsOff = (unsigned)(uintptr_t)&ldsA[buf][aRow[i] * LDA + aKc[i]];
            unsigned long long gaddr =
                (unsigned long long)(uintptr_t)(hs + (size_t)(m0 + aRow[i]) * IN_ + (k0 + aKc[i]));
            asm volatile("global_load_async_to_lds_b128 %0, %1, off"
                         :: "v"(ldsOff), "v"(gaddr) : "memory");
        }
    };

    auto loadW = [&](int kt) {
        const int k0 = kt * BK;
#pragma unroll
        for (int i = 0; i < 4; ++i)
            wStage[i] = *(const uint4*)(wpk + (size_t)(op0 + wRow[i]) * IN_ + (k0 + wKc[i]));
    };

    const __half2 neg1032 = __float2half2_rn(-1032.0f);
    auto stageW = [&](int kt, int buf) {
        const int k0 = kt * BK;
        const int g  = k0 >> 7;
#pragma unroll
        for (int i = 0; i < 4; ++i) {
            const int nlo = wRow[i] * 2;
            const __half2 s2  = *(const __half2*)&ldsS[g * BN + nlo];
            const __half2 sl2 = __half2half2(__low2half(s2));
            const __half2 sh2 = __half2half2(__high2half(s2));
            const __half2* pk2 = (const __half2*)&ldsP[k0 + wKc[i]];
            const __half2 p01 = pk2[0], p23 = pk2[1];
            const __half2 spl01 = __hmul2(sl2, p01), spl23 = __hmul2(sl2, p23);
            const __half2 sph01 = __hmul2(sh2, p01), sph23 = __hmul2(sh2, p23);

            const unsigned v0 = wStage[i].x, v1 = wStage[i].y;
            const unsigned v2 = wStage[i].z, v3 = wStage[i].w;
            __half2 lo01 = h2_from_bits((v0 & 0xFu)        | ((v1 & 0xFu) << 16)        | 0x64006400u);
            __half2 lo23 = h2_from_bits((v2 & 0xFu)        | ((v3 & 0xFu) << 16)        | 0x64006400u);
            __half2 hi01 = h2_from_bits(((v0 >> 4) & 0xFu) | (((v1 >> 4) & 0xFu) << 16) | 0x64006400u);
            __half2 hi23 = h2_from_bits(((v2 >> 4) & 0xFu) | (((v3 >> 4) & 0xFu) << 16) | 0x64006400u);
            lo01 = __hmul2(__hadd2(lo01, neg1032), spl01);
            lo23 = __hmul2(__hadd2(lo23, neg1032), spl23);
            hi01 = __hmul2(__hadd2(hi01, neg1032), sph01);
            hi23 = __hmul2(__hadd2(hi23, neg1032), sph23);

            union { __half2 h2[2]; unsigned long long u; } L, H;
            L.h2[0] = lo01; L.h2[1] = lo23;
            H.h2[0] = hi01; H.h2[1] = hi23;
            *(unsigned long long*)&ldsW[buf][(nlo    ) * LDW + wKc[i]] = L.u;
            *(unsigned long long*)&ldsW[buf][(nlo + 1) * LDW + wKc[i]] = H.u;
        }
    };

    v8f zero = {0.f,0.f,0.f,0.f,0.f,0.f,0.f,0.f};
    v8f acc[4][2];
#pragma unroll
    for (int i = 0; i < 4; ++i)
#pragma unroll
        for (int j = 0; j < 2; ++j) acc[i][j] = zero;

    const int mFragRow = wm * 64 + (lane & 15);
    const int nFragRow = wn * 32 + (lane & 15);
    const int aKoff    = (lane >> 4) << 3;
    const int bKoff    = (lane >> 4) << 3;

    auto compute = [&](int buf) {
#pragma unroll
        for (int ks = 0; ks < 2; ++ks) {
            const int kb = ks * 32;
            v16h af[4];
#pragma unroll
            for (int tm = 0; tm < 4; ++tm) {
                const _Float16* p = &ldsA[buf][(mFragRow + tm * 16) * LDA + kb + aKoff];
                v8h lo = *(const v8h*)p;
                v8h hi = *(const v8h*)(p + 16);
                af[tm] = __builtin_shufflevector(lo, hi,
                         0,1,2,3,4,5,6,7,8,9,10,11,12,13,14,15);
            }
            v16h bf[2];
#pragma unroll
            for (int tn = 0; tn < 2; ++tn) {
                const _Float16* p = &ldsW[buf][(nFragRow + tn * 16) * LDW + kb + bKoff];
                v8h lo = *(const v8h*)p;
                v8h hi = *(const v8h*)(p + 16);
                bf[tn] = __builtin_shufflevector(lo, hi,
                         0,1,2,3,4,5,6,7,8,9,10,11,12,13,14,15);
            }
#pragma unroll
            for (int tm = 0; tm < 4; ++tm)
#pragma unroll
                for (int tn = 0; tn < 2; ++tn)
                    acc[tm][tn] = __builtin_amdgcn_wmma_f32_16x16x32_f16(
                        false, af[tm], false, bf[tn],
                        (short)0, acc[tm][tn], false, false);
        }
    };

    asyncLoadA(0, 0);
#pragma unroll
    for (int i = 0; i < 2; ++i) {
        const int off = (tid + i * 256) * 8;
        *(uint4*)&ldsP[off] = *(const uint4*)(pqs + off);
    }
#pragma unroll
    for (int r = 0; r < 16; ++r) {
        const int idx = tid + r * 256;
        const int n = idx & (BN - 1);
        const int g = idx >> 7;
        ldsS[g * BN + n] = wscale[(size_t)(n0 + n) * NGRP + g];
    }
    loadW(0);
    asm volatile("s_wait_asynccnt 0" ::: "memory");
    __syncthreads();
    stageW(0, 0);
    __syncthreads();

    for (int kt = 0; kt < NKT; ++kt) {
        const int buf = kt & 1;
        if (kt + 1 < NKT) {
            asyncLoadA(kt + 1, buf ^ 1);
            loadW(kt + 1);
        }
        compute(buf);
        if (kt + 1 < NKT) {
            stageW(kt + 1, buf ^ 1);
            asm volatile("s_wait_asynccnt 0" ::: "memory");
        }
        __syncthreads();
    }

    const int rowBase = m0 + wm * 64 + ((lane >> 4) << 3);
    const int colBase = n0 + wn * 32 + (lane & 15);
#pragma unroll
    for (int tm = 0; tm < 4; ++tm)
#pragma unroll
        for (int tn = 0; tn < 2; ++tn)
#pragma unroll
            for (int r = 0; r < 8; ++r)
                out[(size_t)(rowBase + tm * 16 + r) * OUT_ + (colBase + tn * 16)]
                    = acc[tm][tn][r];
}

extern "C" void kernel_launch(void* const* d_in, const int* in_sizes, int n_in,
                              void* d_out, int out_size, void* d_ws, size_t ws_size,
                              hipStream_t stream) {
    (void)in_sizes; (void)n_in; (void)out_size;
    const _Float16* hs     = (const _Float16*)d_in[0];  // [8192, 4096] fp16
    const int*      wpk    = (const int*)d_in[1];       // [6144, 4096] packed int4 pairs
    const _Float16* wscale = (const _Float16*)d_in[2];  // [12288, 32] fp16
    const _Float16* pqs    = (const _Float16*)d_in[3];  // [4096] fp16
    float*          out    = (float*)d_out;             // [8192, 12288]

    dim3 grid(TOKENS_ / BM, OUT_ / BN);                 // 64 x 96 workgroups
    const size_t needWs = (size_t)OUT_ * IN_ * sizeof(_Float16);  // 100 MB
    if (ws_size >= needWs) {
        _Float16* w16 = (_Float16*)d_ws;
        awq_dequant<<<(OUT_ / 2) * (IN_ / 4) / 256, 256, 0, stream>>>(wpk, wscale, pqs, w16);
        gemm_f16_tdm<<<grid, 256, 0, stream>>>(hs, w16, out);
    } else {
        awq_w4a16_wmma_fused<<<grid, 256, 0, stream>>>(hs, wpk, wscale, pqs, out);
    }
}
